// MultiHeadAttention_14525579395763
// MI455X (gfx1250) — matrix operand order repaired and checked by reading, not compiled, not run on hardware
//
#include <hip/hip_runtime.h>

// ---------------------------------------------------------------------------
// MI455X (gfx1250) fused multi-head attention, bf16 WMMA path.
//   P0: one-time fp32 -> bf16 conversion of activations + weights
//   P1: QKV projection  (bf16 WMMA, y = xW^T + b), V stored transposed
//   P2: flash attention, 32 queries/wave (Q.K^T wmma, online softmax, P.V wmma)
//   P3: output projection (bf16 WMMA, fp32 result)
// ---------------------------------------------------------------------------

typedef __attribute__((ext_vector_type(16))) __bf16 v16bf;
typedef __attribute__((ext_vector_type(8)))  __bf16 v8bf;
typedef __attribute__((ext_vector_type(4)))  __bf16 v4bf;
typedef __attribute__((ext_vector_type(8)))  float  v8f;

#define S_LEN   4096
#define D_MODEL 512
#define N_HEADS 8
#define D_K     64
#define BATCH   2

__device__ __forceinline__ v8f bfwmma(v16bf a, v16bf b, v8f c) {
  return __builtin_amdgcn_wmma_f32_16x16x32_bf16(false, a, false, b, (short)0, c,
                                                 false, false);
}

// 16 contiguous bf16 per lane (two b128 loads) -> B fragment
__device__ __forceinline__ v16bf ld_bf16_contig16(const __bf16* p) {
  v8bf a = *(const v8bf*)(p);
  v8bf b = *(const v8bf*)(p + 8);
  v16bf r;
#pragma unroll
  for (int i = 0; i < 8; ++i) { r[i] = a[i]; r[i + 8] = b[i]; }
  return r;
}

// A-fragment per-lane layout: K chunks {p[0..7], p[8..15]}
__device__ __forceinline__ v16bf ld_bf16_Afrag(const __bf16* p) {
  v8bf a = *(const v8bf*)(p);
  v8bf b = *(const v8bf*)(p + 8);
  v16bf r;
#pragma unroll
  for (int i = 0; i < 8; ++i) { r[i] = a[i]; r[i + 8] = b[i]; }
  return r;
}

// ---------------------------------------------------------------------------
// P0: one-time fp32 -> bf16 conversion (vectorized).  n must be multiple of 4.
// ---------------------------------------------------------------------------
__global__ __launch_bounds__(256) void cvt_f32_to_bf16(
    const float* __restrict__ src, __bf16* __restrict__ dst, int n) {
  const int i = (blockIdx.x * 256 + threadIdx.x) * 4;
  if (i < n) {
    const float4 f = *(const float4*)(src + i);
    v4bf o;
    o[0] = (__bf16)f.x; o[1] = (__bf16)f.y; o[2] = (__bf16)f.z; o[3] = (__bf16)f.w;
    *(v4bf*)(dst + i) = o;
  }
}

// ---------------------------------------------------------------------------
// P1: QKV projection.  grid = (BS/16, 1, 3), block = 256 (8 waves).
// Wave w owns head w (64 columns = 4 N-tiles) of one 16-row M-tile:
// A fragment loaded once per K-step, reused across 4 WMMAs.
// z=0 -> Q [B,H,S,dk], z=1 -> K [B,H,S,dk], z=2 -> V transposed [B,H,dk,S]
// ---------------------------------------------------------------------------
__global__ __launch_bounds__(256) void mha_qkv_proj(
    const __bf16* __restrict__ Xq, const __bf16* __restrict__ Xk,
    const __bf16* __restrict__ Xv,
    const __bf16* __restrict__ Wqb, const float* __restrict__ bq,
    const __bf16* __restrict__ Wkb, const float* __restrict__ bk,
    const __bf16* __restrict__ Wvb, const float* __restrict__ bv,
    __bf16* __restrict__ Qh, __bf16* __restrict__ Kh, __bf16* __restrict__ Vt) {
  const int l    = threadIdx.x & 31;          // lane in wave
  const int wave = threadIdx.x >> 5;          // 0..7 == head
  const int half = l >> 4;
  const int mb   = blockIdx.x * 16;
  const int nb   = wave * D_K;                // 4 n-tiles: nb .. nb+63

  const __bf16* X; const __bf16* W; const float* bias; __bf16* out; bool tr;
  if (blockIdx.z == 0)      { X = Xq; W = Wqb; bias = bq; out = Qh; tr = false; }
  else if (blockIdx.z == 1) { X = Xk; W = Wkb; bias = bk; out = Kh; tr = false; }
  else                      { X = Xv; W = Wvb; bias = bv; out = Vt; tr = true;  }

  const int arow = mb + (l & 15);

  v8f acc[4];
#pragma unroll
  for (int t = 0; t < 4; ++t)
#pragma unroll
    for (int i = 0; i < 8; ++i) acc[t][i] = 0.0f;

#pragma unroll 4
  for (int kk = 0; kk < D_MODEL; kk += 32) {
    const v16bf a = ld_bf16_Afrag(X + arow * D_MODEL + kk + half * 16);
#pragma unroll
    for (int t = 0; t < 4; ++t) {
      const int ncol = nb + t * 16 + (l & 15);   // B lane -> N col
      const v16bf b = ld_bf16_contig16(W + ncol * D_MODEL + kk + half * 16);
      acc[t] = bfwmma(a, b, acc[t]);
    }
  }

  const int h = wave;
#pragma unroll
  for (int t = 0; t < 4; ++t) {
    const int d  = t * 16 + (l & 15);
    const float bn = bias[nb + t * 16 + (l & 15)];
#pragma unroll
    for (int i = 0; i < 8; ++i) {
      const int m  = mb + i + half * 8;         // row in [0, B*S)
      const int bb = m >> 12;                   // / 4096
      const int ss = m & (S_LEN - 1);
      const float val = acc[t][i] + bn;
      if (!tr)
        out[(((bb * N_HEADS + h) * S_LEN) + ss) * D_K + d] = (__bf16)val;
      else
        out[(((bb * N_HEADS + h) * D_K) + d) * S_LEN + ss] = (__bf16)val;
    }
  }
}

// ---------------------------------------------------------------------------
// P2: causal flash attention.  grid = (S/32, H, B), block = 32 (one wave).
// 32 queries per wave (two 16-row M-tiles held in registers); key blocks of
// 32 (two 16-wide score tiles).  K and V fragments are loaded once and reused
// by both M-tiles -> 16 WMMAs per key block.  Writes context as bf16.
// ---------------------------------------------------------------------------
__global__ __launch_bounds__(32) void mha_flash(
    const __bf16* __restrict__ Qh, const __bf16* __restrict__ Kh,
    const __bf16* __restrict__ Vt, __bf16* __restrict__ ctx) {
  __shared__ alignas(16) __bf16 Plds[32][32];   // C-layout -> A-layout staging

  const int l    = threadIdx.x;
  const int half = l >> 4;
  const int qb   = blockIdx.x * 32;
  const int h    = blockIdx.y;
  const int b    = blockIdx.z;

  const __bf16* Q = Qh + (size_t)(b * N_HEADS + h) * S_LEN * D_K;
  const __bf16* K = Kh + (size_t)(b * N_HEADS + h) * S_LEN * D_K;
  const __bf16* V = Vt + (size_t)(b * N_HEADS + h) * D_K * S_LEN;

  // Q A-fragments: 2 M-tiles x 2 K-steps (dk = 64), reused across key loop
  v16bf qa[2][2];
#pragma unroll
  for (int r = 0; r < 2; ++r) {
    const int mQ = qb + r * 16 + (l & 15);
    qa[r][0] = ld_bf16_Afrag(Q + mQ * D_K + 0 + half * 16);
    qa[r][1] = ld_bf16_Afrag(Q + mQ * D_K + 32 + half * 16);
  }

  v8f cacc[2][4];
  float mrow[2][8], lrow[2][8];
#pragma unroll
  for (int r = 0; r < 2; ++r) {
#pragma unroll
    for (int t = 0; t < 4; ++t)
#pragma unroll
      for (int i = 0; i < 8; ++i) cacc[r][t][i] = 0.0f;
#pragma unroll
    for (int i = 0; i < 8; ++i) { mrow[r][i] = -3.0e38f; lrow[r][i] = 0.0f; }
  }

  const float scale = 0.125f;  // 1/sqrt(64)
  const int kmax = qb + 31;    // last valid (causal) key for this q block

  for (int kb = 0; kb <= kmax; kb += 32) {
    if (kb + 32 <= kmax)
      __builtin_prefetch((const void*)(K + (kb + 32 + (l & 15)) * D_K), 0, 1);

    // ---- scores: 2 key tiles x 2 M-tiles; K frags loaded once per key tile
    v8f s[2][2];                               // [mtile][keytile]
#pragma unroll
    for (int r = 0; r < 2; ++r)
#pragma unroll
      for (int kt = 0; kt < 2; ++kt)
#pragma unroll
        for (int i = 0; i < 8; ++i) s[r][kt][i] = 0.0f;

#pragma unroll
    for (int kt = 0; kt < 2; ++kt) {
      const int key = kb + kt * 16 + (l & 15);   // B lane -> key
      const v16bf kb0 = ld_bf16_contig16(K + key * D_K + 0 + half * 16);
      const v16bf kb1 = ld_bf16_contig16(K + key * D_K + 32 + half * 16);
#pragma unroll
      for (int r = 0; r < 2; ++r) {
        s[r][kt] = bfwmma(qa[r][0], kb0, s[r][kt]);
        s[r][kt] = bfwmma(qa[r][1], kb1, s[r][kt]);
      }
    }

    // ---- per-M-tile: causal mask + scale + online softmax ----
#pragma unroll
    for (int r = 0; r < 2; ++r) {
      const int key0 = kb + (l & 15);
      float rmax[8];
#pragma unroll
      for (int i = 0; i < 8; ++i) {
        const int mg = qb + r * 16 + i + half * 8;   // global query row
        const float a0 = (key0 <= mg)      ? s[r][0][i] * scale : -3.0e38f;
        const float a1 = (key0 + 16 <= mg) ? s[r][1][i] * scale : -3.0e38f;
        s[r][0][i] = a0; s[r][1][i] = a1;
        rmax[i] = fmaxf(a0, a1);
      }
      // row max across keys: reduce over each 16-lane half
#pragma unroll
      for (int off = 1; off < 16; off <<= 1)
#pragma unroll
        for (int i = 0; i < 8; ++i)
          rmax[i] = fmaxf(rmax[i], __shfl_xor(rmax[i], off, 32));

      float alpha[8], rsum[8];
#pragma unroll
      for (int i = 0; i < 8; ++i) {
        const float mn = fmaxf(mrow[r][i], rmax[i]);
        alpha[i] = __expf(mrow[r][i] - mn);
        mrow[r][i] = mn;
        const float p0 = __expf(s[r][0][i] - mn);  // masked -> underflow to 0
        const float p1 = __expf(s[r][1][i] - mn);
        s[r][0][i] = p0; s[r][1][i] = p1;
        rsum[i] = p0 + p1;
      }
#pragma unroll
      for (int off = 1; off < 16; off <<= 1)
#pragma unroll
        for (int i = 0; i < 8; ++i)
          rsum[i] += __shfl_xor(rsum[i], off, 32);
#pragma unroll
      for (int i = 0; i < 8; ++i) lrow[r][i] = lrow[r][i] * alpha[i] + rsum[i];
#pragma unroll
      for (int t = 0; t < 4; ++t)
#pragma unroll
        for (int i = 0; i < 8; ++i) cacc[r][t][i] *= alpha[i];
    }

    // ---- P: C-layout -> A-layout via LDS (both M-tiles) ----
    __syncthreads();
#pragma unroll
    for (int r = 0; r < 2; ++r)
#pragma unroll
      for (int i = 0; i < 8; ++i) {
        Plds[r * 16 + half * 8 + i][(l & 15)]      = (__bf16)s[r][0][i];
        Plds[r * 16 + half * 8 + i][16 + (l & 15)] = (__bf16)s[r][1][i];
      }
    __syncthreads();
    v16bf pa[2];
#pragma unroll
    for (int r = 0; r < 2; ++r)
      pa[r] = ld_bf16_Afrag(&Plds[r * 16 + (l & 15)][half * 16]);

    // ---- context += P(16x32).V(32x64); V frag shared by both M-tiles ----
#pragma unroll
    for (int t = 0; t < 4; ++t) {
      const int d = t * 16 + (l & 15);
      const v16bf vb = ld_bf16_contig16(V + d * S_LEN + kb + half * 16);
#pragma unroll
      for (int r = 0; r < 2; ++r)
        cacc[r][t] = bfwmma(pa[r], vb, cacc[r][t]);
    }
  }

  // normalize + write context (bf16, [B,S,D], heads re-interleaved)
#pragma unroll
  for (int r = 0; r < 2; ++r)
#pragma unroll
    for (int t = 0; t < 4; ++t) {
      const int d = h * D_K + t * 16 + (l & 15);
#pragma unroll
      for (int i = 0; i < 8; ++i) {
        const int m = qb + r * 16 + i + half * 8;
        ctx[(size_t)(b * S_LEN + m) * D_MODEL + d] =
            (__bf16)(cacc[r][t][i] / lrow[r][i]);
      }
    }
}

// ---------------------------------------------------------------------------
// P3: output projection.  grid = (BS/16), block = 256 (8 waves).
// Wave w owns columns [w*64, w*64+64): A fragment reused across 4 WMMAs.
// ---------------------------------------------------------------------------
__global__ __launch_bounds__(256) void mha_out_proj(
    const __bf16* __restrict__ ctxb, const __bf16* __restrict__ Wob,
    const float* __restrict__ bo, float* __restrict__ outp) {
  const int l    = threadIdx.x & 31;
  const int wave = threadIdx.x >> 5;
  const int half = l >> 4;
  const int mb   = blockIdx.x * 16;
  const int nb   = wave * D_K;
  const int arow = mb + (l & 15);

  v8f acc[4];
#pragma unroll
  for (int t = 0; t < 4; ++t)
#pragma unroll
    for (int i = 0; i < 8; ++i) acc[t][i] = 0.0f;

#pragma unroll 4
  for (int kk = 0; kk < D_MODEL; kk += 32) {
    const v16bf a = ld_bf16_Afrag(ctxb + arow * D_MODEL + kk + half * 16);
#pragma unroll
    for (int t = 0; t < 4; ++t) {
      const int ncol = nb + t * 16 + (l & 15);
      const v16bf b = ld_bf16_contig16(Wob + ncol * D_MODEL + kk + half * 16);
      acc[t] = bfwmma(a, b, acc[t]);
    }
  }

#pragma unroll
  for (int t = 0; t < 4; ++t) {
    const int ncol = nb + t * 16 + (l & 15);
    const float bn = bo[ncol];
#pragma unroll
    for (int i = 0; i < 8; ++i) {
      const int m = mb + i + half * 8;
      outp[(size_t)m * D_MODEL + ncol] = acc[t][i] + bn;
    }
  }
}

// ---------------------------------------------------------------------------
extern "C" void kernel_launch(void* const* d_in, const int* in_sizes, int n_in,
                              void* d_out, int out_size, void* d_ws, size_t ws_size,
                              hipStream_t stream) {
  (void)in_sizes; (void)n_in; (void)out_size; (void)ws_size;
  const float* q  = (const float*)d_in[0];
  const float* k  = (const float*)d_in[1];
  const float* v  = (const float*)d_in[2];
  // d_in[3] = causal mask: computed analytically, never read (saves 16 MB)
  const float* Wq = (const float*)d_in[4];
  const float* bq = (const float*)d_in[5];
  const float* Wk = (const float*)d_in[6];
  const float* bk = (const float*)d_in[7];
  const float* Wv = (const float*)d_in[8];
  const float* bv = (const float*)d_in[9];
  const float* Wo = (const float*)d_in[10];
  const float* bo = (const float*)d_in[11];

  // workspace layout (bf16 elements):
  //   Xq | Xk | Xv            : BATCH*S*D each        (4M elems = 8MB each)
  //   Wqb | Wkb | Wvb | Wob   : D*D each              (256K elems = 512KB each)
  //   Qh | Kh | Vt            : BATCH*H*S*dk each     (4M elems = 8MB each)
  //   ctxb                    : BATCH*S*D             (4M elems = 8MB)
  char* ws = (char*)d_ws;
  const size_t nX = (size_t)BATCH * S_LEN * D_MODEL;   // 4,194,304
  const size_t nW = (size_t)D_MODEL * D_MODEL;         // 262,144
  __bf16* Xq   = (__bf16*)(ws);
  __bf16* Xk   = Xq  + nX;
  __bf16* Xv   = Xk  + nX;
  __bf16* Wqb  = Xv  + nX;
  __bf16* Wkb  = Wqb + nW;
  __bf16* Wvb  = Wkb + nW;
  __bf16* Wob  = Wvb + nW;
  __bf16* Qh   = Wob + nW;
  __bf16* Kh   = Qh  + nX;
  __bf16* Vt   = Kh  + nX;
  __bf16* ctxb = Vt  + nX;

  // P0: one-time conversions
  const int gx = (int)(nX / 4 / 256);   // 4096 blocks
  const int gw = (int)(nW / 4 / 256);   // 256 blocks
  cvt_f32_to_bf16<<<gx, 256, 0, stream>>>(q,  Xq,  (int)nX);
  cvt_f32_to_bf16<<<gx, 256, 0, stream>>>(k,  Xk,  (int)nX);
  cvt_f32_to_bf16<<<gx, 256, 0, stream>>>(v,  Xv,  (int)nX);
  cvt_f32_to_bf16<<<gw, 256, 0, stream>>>(Wq, Wqb, (int)nW);
  cvt_f32_to_bf16<<<gw, 256, 0, stream>>>(Wk, Wkb, (int)nW);
  cvt_f32_to_bf16<<<gw, 256, 0, stream>>>(Wv, Wvb, (int)nW);
  cvt_f32_to_bf16<<<gw, 256, 0, stream>>>(Wo, Wob, (int)nW);

  const int mtiles = (BATCH * S_LEN) / 16;  // 512

  // P1: QKV projections (bf16 in, bf16 out)
  mha_qkv_proj<<<dim3(mtiles, 1, 3), dim3(256), 0, stream>>>(
      Xq, Xk, Xv, Wqb, bq, Wkb, bk, Wvb, bv, Qh, Kh, Vt);

  // P2: flash attention (32 queries per wave)
  mha_flash<<<dim3(S_LEN / 32, N_HEADS, BATCH), dim3(32), 0, stream>>>(
      Qh, Kh, Vt, ctxb);

  // P3: output projection (fp32 result)
  mha_out_proj<<<dim3(mtiles), dim3(256), 0, stream>>>(
      ctxb, Wob, bo, (float*)d_out);
}
